// Net_85959475462272
// MI455X (gfx1250) — matrix operand order repaired and checked by reading, not compiled, not run on hardware
//
#include <hip/hip_runtime.h>
#include <hip/hip_bf16.h>
#include <stdint.h>

// ---------------------------------------------------------------------------
// SNN forward (snnTorch-style LIF, 2 layers) for MI455X / gfx1250.
//
// Kernel 1: cur1[t*B+b][h] = x[t][b][:] @ W1^T + b1      (bf16 WMMA, f32 acc)
//   - memory-bound: streams x (474 MB) exactly once;
//   - 256-row M tile per block, wave owns 2 M-tiles x 8 N-tiles (16 accums)
//     so every B fragment load feeds two v_wmma;
//   - K tail (1156 = 36*32 + 4) peeled so main chunks use pure b128 loads;
//   - hardware f32->bf16 conversion (native casts -> packed cvt ops).
// Kernel 2: per-block time loop over 32 batch rows (rows are independent):
//   mem1 = 0.9*mem1 + cur1 + spk1@V1^T + bV1 - reset ; spk1 = (mem1>1)
//   spk1@V1^T via bf16 WMMA (spikes exact in bf16); V1 in LDS (32 KB);
//   mem1 lives in WMMA accumulator registers across all 50 steps;
//   layer2 (128->10) via VALU dots; spk2 -> d_out.
// ---------------------------------------------------------------------------

typedef __bf16 bf16_t;
typedef bf16_t v16bf __attribute__((ext_vector_type(16)));
typedef bf16_t v2bf  __attribute__((ext_vector_type(2)));
typedef float  v8f   __attribute__((ext_vector_type(8)));

struct Frag32B { uint4 lo, hi; };

// hardware f32->bf16 (RNE) — compiler emits packed cvt where possible
__device__ __forceinline__ unsigned int cvt2(float a, float b) {
  v2bf r; r.x = (bf16_t)a; r.y = (bf16_t)b;
  return __builtin_bit_cast(unsigned int, r);
}
__device__ __forceinline__ unsigned short bf16bits(float f) {
  bf16_t h = (bf16_t)f;
  return __builtin_bit_cast(unsigned short, h);
}
__device__ __forceinline__ float bf2f(unsigned short h) {
  unsigned int u = ((unsigned int)h) << 16;
  return __builtin_bit_cast(float, u);
}
// Build a 16xbf16 WMMA fragment from two 16B LDS reads (ds_load_b128 x2).
__device__ __forceinline__ v16bf load_frag(const unsigned short* base,
                                           int off_lo, int off_hi) {
  Frag32B f;
  f.lo = *(const uint4*)(base + off_lo);
  f.hi = *(const uint4*)(base + off_hi);
  return __builtin_bit_cast(v16bf, f);
}
__device__ __forceinline__ v8f wmma_bf16(v16bf a, v16bf b, v8f c) {
  return __builtin_amdgcn_wmma_f32_16x16x32_bf16(false, a, false, b, (short)0,
                                                 c, false, false);
}

#define K_DIM   1156
#define KCHUNKS 37          // 36 full 32-chunks + 1 padded tail (k 1152..1155)
#define N_DIM   128
#define T_STEPS 50
#define BATCH   2048
#define HD      128
#define OUTD    10
#define ROWS    32          // batch rows per block in kernel 2
#define BLOCK_M 256         // M rows per block in kernel 1

// ============================ Kernel 1 ====================================
// Grid: 400 blocks (M = T*B = 102400). Block: 256 thr (8 waves).
// Wave w owns M-tiles {2w, 2w+1} (32 rows) x all 8 N-tiles.
__global__ __launch_bounds__(256) void gemm_xw1_kernel(
    const float* __restrict__ x, const float* __restrict__ W1,
    const float* __restrict__ b1, float* __restrict__ cur1) {
  __shared__ __align__(16) unsigned short aS[BLOCK_M * 32]; // x tile,  16 KB
  __shared__ __align__(16) unsigned short bS[128 * 32];     // W1 tile, 8 KB

  const int tid  = threadIdx.x;
  const int wave = tid >> 5, lane = tid & 31;
  const int l16  = lane & 15;
  const int hi   = lane >> 4;
  const size_t blockM = (size_t)blockIdx.x * BLOCK_M;

  v8f acc[2][8] = {};

  // loaders: A -> thread owns one row (32 k per chunk, 8x float4);
  //          B -> 2 threads per W1 row (16 k each, 4x float4)
  const float* xrow = x + (blockM + tid) * K_DIM;
  const int bn = tid >> 1, bk0 = (tid & 1) * 16;
  const float* wrow = W1 + (size_t)bn * K_DIM + bk0;

  const int a_lo = hi ? 8 : 0,  a_hi = hi ? 24 : 16;  // A frag K offsets
  const int b_lo = hi ? 8 : 0,  b_hi = hi ? 24 : 16;  // B frag K offsets

  for (int kc = 0; kc < KCHUNKS; ++kc) {
    const int kbase = kc * 32;
    if (kc < KCHUNKS - 1) {
#pragma unroll
      for (int i = 0; i < 8; ++i) {           // A: global_load_b128 x8
        const float4 q = *(const float4*)(xrow + kbase + i * 4);
        uint2 u; u.x = cvt2(q.x, q.y); u.y = cvt2(q.z, q.w);
        *(uint2*)&aS[tid * 32 + i * 4] = u;
      }
#pragma unroll
      for (int i = 0; i < 4; ++i) {           // B: global_load_b128 x4
        const float4 q = *(const float4*)(wrow + kbase + i * 4);
        uint2 u; u.x = cvt2(q.x, q.y); u.y = cvt2(q.z, q.w);
        *(uint2*)&bS[bn * 32 + bk0 + i * 4] = u;
      }
      __builtin_prefetch(xrow + kbase + 32, 0, 3);   // next chunk of x
    } else {
      // tail chunk: only k 1152..1155 valid; zero-pad rest of the 32-chunk
      uint2 z; z.x = 0u; z.y = 0u;
      {
        const float4 q = *(const float4*)(xrow + kbase);
        uint2 u; u.x = cvt2(q.x, q.y); u.y = cvt2(q.z, q.w);
        *(uint2*)&aS[tid * 32] = u;
#pragma unroll
        for (int i = 1; i < 8; ++i) *(uint2*)&aS[tid * 32 + i * 4] = z;
      }
      if (bk0 == 0) {
        const float4 q = *(const float4*)(wrow + kbase);
        uint2 u; u.x = cvt2(q.x, q.y); u.y = cvt2(q.z, q.w);
        *(uint2*)&bS[bn * 32] = u;
#pragma unroll
        for (int i = 1; i < 4; ++i) *(uint2*)&bS[bn * 32 + i * 4] = z;
      } else {
#pragma unroll
        for (int i = 0; i < 4; ++i) *(uint2*)&bS[bn * 32 + bk0 + i * 4] = z;
      }
    }
    __syncthreads();

    // A frags (16x32 layout: lane<16 K{0..7,16..23}; lane>=16 K{8..15,24..31})
    const unsigned short* ar0 = &aS[(wave * 32 + l16) * 32];
    const unsigned short* ar1 = &aS[(wave * 32 + 16 + l16) * 32];
    const v16bf a0 = load_frag(ar0, a_lo, a_hi);
    const v16bf a1 = load_frag(ar1, a_lo, a_hi);
#pragma unroll
    for (int nt = 0; nt < 8; ++nt) {
      // B frag (32x16: lane<16 K{0..7,16..23}, lane>=16 K{8..15,24..31}, col = lane%16)
      const unsigned short* brow = &bS[(nt * 16 + l16) * 32];
      const v16bf bv = load_frag(brow, b_lo, b_hi);
      acc[0][nt] = wmma_bf16(a0, bv, acc[0][nt]);
      acc[1][nt] = wmma_bf16(a1, bv, acc[1][nt]);
    }
    __syncthreads();
  }

  // D layout: VGPR v -> (m = v + 8*hi, n = lane%16) within each 16x16 tile.
#pragma unroll
  for (int mt = 0; mt < 2; ++mt) {
#pragma unroll
    for (int nt = 0; nt < 8; ++nt) {
      const int ncol = nt * 16 + l16;
      const float bias = b1[ncol];
#pragma unroll
      for (int v = 0; v < 8; ++v) {
        const int m = wave * 32 + mt * 16 + v + hi * 8;
        cur1[(blockM + m) * N_DIM + ncol] = acc[mt][nt][v] + bias;
      }
    }
  }
}

// ============================ Kernel 2 ====================================
// Grid: 64 blocks x 256 thr (8 waves). Block handles 32 batch rows, all T.
// Wave w: M-tile mt = w>>2 (rows mt*16..+15), N-tiles {2*(w&3), 2*(w&3)+1}.
__global__ __launch_bounds__(256) void snn_recurrent_kernel(
    const float* __restrict__ cur1, const float* __restrict__ V1,
    const float* __restrict__ bV1,  const float* __restrict__ W2,
    const float* __restrict__ b2,   const float* __restrict__ V2,
    const float* __restrict__ bV2,  float* __restrict__ out) {
  __shared__ __align__(16) unsigned short v1S[HD * HD];    // V1 [n][k] bf16 32KB
  __shared__ __align__(16) unsigned short spkS[ROWS * HD]; // spk1 [r][h] bf16
  __shared__ float w2S[OUTD * HD];
  __shared__ float v2S[OUTD * OUTD];
  __shared__ float b2S[OUTD], bv2S[OUTD];
  __shared__ float spk2S[ROWS * OUTD];
  __shared__ float mem2S[ROWS * OUTD];

  const int tid  = threadIdx.x;
  const int wave = tid >> 5, lane = tid & 31;
  const int l16  = lane & 15, hi = lane >> 4;
  const int mt   = wave >> 2;            // 0..1
  const int np   = wave & 3;             // 0..3
  const int rowBase = blockIdx.x * ROWS;

  for (int i = tid; i < HD * HD; i += 256)     v1S[i] = bf16bits(V1[i]);
  for (int i = tid; i < OUTD * HD; i += 256)   w2S[i] = W2[i];
  for (int i = tid; i < OUTD * OUTD; i += 256) v2S[i] = V2[i];
  if (tid < OUTD) { b2S[tid] = b2[tid]; bv2S[tid] = bV2[tid]; }
  for (int i = tid; i < ROWS * HD; i += 256)   spkS[i] = 0;
  for (int i = tid; i < ROWS * OUTD; i += 256) { spk2S[i] = 0.f; mem2S[i] = 0.f; }

  v8f mem1[2] = {};                     // LIF membrane state, lives in VGPRs
  float bv1v[2]; int ncol[2];
#pragma unroll
  for (int j = 0; j < 2; ++j) {
    ncol[j] = (np * 2 + j) * 16 + l16;
    bv1v[j] = bV1[ncol[j]];
  }
  const int rloc0 = mt * 16 + hi * 8;   // local row of accum element v is rloc0+v
  const int a_lo = hi ? 8 : 0,  a_hi = hi ? 24 : 16;
  const int b_lo = hi ? 8 : 0,  b_hi = hi ? 24 : 16;

  __syncthreads();

  for (int t = 0; t < T_STEPS; ++t) {
    // ---- phase 1: rec = spk1(prev) @ V1^T for this wave's tiles (WMMA) ----
    v8f rec[2] = {};
#pragma unroll
    for (int kc = 0; kc < 4; ++kc) {
      const int kb = kc * 32;
      const unsigned short* arow = &spkS[(mt * 16 + l16) * HD + kb];
      const v16bf av = load_frag(arow, a_lo, a_hi);
#pragma unroll
      for (int j = 0; j < 2; ++j) {
        const unsigned short* brow = &v1S[ncol[j] * HD + kb];
        const v16bf bv = load_frag(brow, b_lo, b_hi);
        rec[j] = wmma_bf16(av, bv, rec[j]);
      }
    }
    // precomputed feed-forward current for this wave's (m,n) elements
    float cv[2][8];
#pragma unroll
    for (int j = 0; j < 2; ++j)
#pragma unroll
      for (int v = 0; v < 8; ++v)
        cv[j][v] = cur1[((size_t)t * BATCH + rowBase + rloc0 + v) * HD + ncol[j]];
    __syncthreads();

    // ---- phase 2: membrane update + spike, write spk1(t) to LDS ----------
#pragma unroll
    for (int j = 0; j < 2; ++j) {
#pragma unroll
      for (int v = 0; v < 8; ++v) {
        float m1 = mem1[j][v];
        const float rst = (m1 > 1.0f) ? 1.0f : 0.0f;   // reset from prev mem
        m1 = 0.9f * m1 + cv[j][v] + rec[j][v] + bv1v[j] - rst;
        mem1[j][v] = m1;
        spkS[(rloc0 + v) * HD + ncol[j]] =
            (m1 > 1.0f) ? (unsigned short)0x3F80 : (unsigned short)0;  // bf16 1/0
      }
    }
    __syncthreads();

    // ---- phase 3: layer 2 (320 dots of length 128) ------------------------
    float nm[2], ns[2];
#pragma unroll
    for (int rep = 0; rep < 2; ++rep) {
      const int oi = tid + rep * 256;
      if (oi < ROWS * OUTD) {
        const int r = oi / OUTD, o = oi - r * OUTD;
        float dot = b2S[o] + bv2S[o];
        const unsigned short* sr = &spkS[r * HD];
        const float* wr = &w2S[o * HD];
#pragma unroll 8
        for (int h = 0; h < HD; ++h) dot += bf2f(sr[h]) * wr[h];
#pragma unroll
        for (int oo = 0; oo < OUTD; ++oo)
          dot += spk2S[r * OUTD + oo] * v2S[o * OUTD + oo];
        float m2 = mem2S[oi];
        const float rst2 = (m2 > 1.0f) ? 1.0f : 0.0f;
        m2 = 0.9f * m2 + dot - rst2;
        nm[rep] = m2;
        ns[rep] = (m2 > 1.0f) ? 1.0f : 0.0f;
      }
    }
    __syncthreads();

    // ---- phase 4: commit state + emit spk2 --------------------------------
#pragma unroll
    for (int rep = 0; rep < 2; ++rep) {
      const int oi = tid + rep * 256;
      if (oi < ROWS * OUTD) {
        const int r = oi / OUTD, o = oi - r * OUTD;
        mem2S[oi] = nm[rep];
        spk2S[oi] = ns[rep];
        out[((size_t)t * BATCH + rowBase + r) * OUTD + o] = ns[rep];
      }
    }
    __syncthreads();
  }
}

// ============================ Launch ======================================
extern "C" void kernel_launch(void* const* d_in, const int* in_sizes, int n_in,
                              void* d_out, int out_size, void* d_ws, size_t ws_size,
                              hipStream_t stream) {
  const float* x   = (const float*)d_in[0];
  const float* W1  = (const float*)d_in[1];
  const float* b1  = (const float*)d_in[2];
  const float* V1  = (const float*)d_in[3];
  const float* bV1 = (const float*)d_in[4];
  const float* W2  = (const float*)d_in[5];
  const float* b2  = (const float*)d_in[6];
  const float* V2  = (const float*)d_in[7];
  const float* bV2 = (const float*)d_in[8];
  float* out  = (float*)d_out;
  float* cur1 = (float*)d_ws;     // needs T*B*H*4 = 52,428,800 bytes

  // Phase A: time-independent big GEMM (memory-bound on x, bf16 WMMA).
  gemm_xw1_kernel<<<(T_STEPS * BATCH) / BLOCK_M, 256, 0, stream>>>(x, W1, b1, cur1);
  // Phase B: serial-in-T recurrent dynamics; batch-parallel across blocks.
  snn_recurrent_kernel<<<BATCH / ROWS, 256, 0, stream>>>(cur1, V1, bV1, W2, b2, V2, bV2, out);
}
